// AttentionCT_19353122636294
// MI455X (gfx1250) — matrix operand order repaired and checked by reading, not compiled, not run on hardware
//
#include <hip/hip_runtime.h>
#include <hip/hip_bf16.h>

// Problem constants (from reference)
#define BATCH   8
#define CDIM    192
#define HW      16384
#define HEADS   8
#define CT      24      // channels per head
#define OC3     576     // 3*CDIM
#define EPSN    1e-12f

typedef __attribute__((ext_vector_type(16))) __bf16 v16bf;
typedef __attribute__((ext_vector_type(8)))  float  v8f;

union BF16x16 { uint4 q[2]; v16bf v; };

__device__ __forceinline__ unsigned short f32_to_bf16(float f) {
    union { float f; unsigned int u; } v; v.f = f;
    unsigned int u = v.u;
    u += 0x7FFFu + ((u >> 16) & 1u);   // round-to-nearest-even
    return (unsigned short)(u >> 16);
}
__device__ __forceinline__ float bf16_to_f32(unsigned short h) {
    union { unsigned int u; float f; } v; v.u = ((unsigned int)h) << 16;
    return v.f;
}
__device__ __forceinline__ uint4 pack8_bf16(v8f a) {
    union { unsigned short s[8]; uint4 q; } u;
#pragma unroll
    for (int i = 0; i < 8; ++i) u.s[i] = f32_to_bf16(a[i]);
    return u.q;
}
__device__ __forceinline__ v8f wmma_bf16(v16bf a, v16bf b, v8f c) {
    // D = A(16x32 bf16) * B(32x16 bf16) + C(f32)
    return __builtin_amdgcn_wmma_f32_16x16x32_bf16(false, a, false, b,
                                                   (short)0, c, false, false);
}
__device__ __forceinline__ v8f v8f_zero() {
    v8f z = {0.f,0.f,0.f,0.f,0.f,0.f,0.f,0.f};
    return z;
}

// ---------------------------------------------------------------- weights f32->bf16
__global__ void k_convert_w(const float* __restrict__ qkv_w,
                            const float* __restrict__ proj_w,
                            unsigned short* __restrict__ wq,
                            unsigned short* __restrict__ wp) {
    int i = blockIdx.x * 256 + threadIdx.x;
    if (i < OC3 * CDIM)  wq[i] = f32_to_bf16(qkv_w[i]);
    if (i < CDIM * CDIM) wp[i] = f32_to_bf16(proj_w[i]);
}

// ------------------------------------------------- depthwise 3x3 -> y_t[b][n][c] bf16
// Block: 32 channels x 64 pixels; LDS transpose so reads coalesce over n and
// writes coalesce over c.
__global__ void k_dwconv(const float* __restrict__ x,
                         const float* __restrict__ dww,
                         unsigned short* __restrict__ y_t) {
    __shared__ unsigned short tile[32][66];   // 66 stride -> conflict-free transpose
    const int nblk = blockIdx.x, cblk = blockIdx.y, b = blockIdx.z;
    const int tid = threadIdx.x;
#pragma unroll
    for (int it = 0; it < 8; ++it) {
        int idx = it * 256 + tid;
        int c_l = idx >> 6, n_l = idx & 63;
        int c = cblk * 32 + c_l;
        int n = nblk * 64 + n_l;
        int hh = n >> 7, ww = n & 127;
        const float* xp = x + ((size_t)(b * CDIM + c) << 14);
        const float* wk = dww + c * 9;
        float acc = 0.f;
#pragma unroll
        for (int dh = -1; dh <= 1; ++dh) {
            int h2 = hh + dh;
            if (h2 < 0 || h2 > 127) continue;
#pragma unroll
            for (int dw = -1; dw <= 1; ++dw) {
                int w2 = ww + dw;
                if (w2 < 0 || w2 > 127) continue;
                acc += wk[(dh + 1) * 3 + (dw + 1)] * xp[h2 * 128 + w2];
            }
        }
        tile[c_l][n_l] = f32_to_bf16(acc);
    }
    __syncthreads();
#pragma unroll
    for (int it = 0; it < 8; ++it) {
        int idx = it * 256 + tid;
        int n_l = idx >> 5, c_l = idx & 31;
        y_t[((size_t)(b * HW + nblk * 64 + n_l)) * CDIM + cblk * 32 + c_l] = tile[c_l][n_l];
    }
}

// ----------------------------------------------------- qkv GEMM: [576x192]x[192xN]
// One wave = 16 out-channels x 64 pixels, K=192 in 6 WMMA steps.
// q,k rows (o<384) stored [b][o][n]; v rows stored transposed v_t[b][n][c].
__global__ void k_gemm_qkv(const unsigned short* __restrict__ wq,
                           const unsigned short* __restrict__ y_t,
                           unsigned short* __restrict__ qk,
                           unsigned short* __restrict__ v_t) {
    const int wid  = blockIdx.x * 8 + (threadIdx.x >> 5);
    const int lane = threadIdx.x & 31;
    const int n64   = wid & 255;
    const int mtile = (wid >> 8) % 36;
    const int b     = wid / (256 * 36);
    const int hi = lane >> 4, l15 = lane & 15;
    const int o_row = mtile * 16 + l15;
    const int n0 = n64 * 64;

    v8f acc[4];
#pragma unroll
    for (int t = 0; t < 4; ++t) acc[t] = v8f_zero();

    for (int kk = 0; kk < 6; ++kk) {
        const int c0 = kk * 32;
        BF16x16 A;
        const unsigned short* ap = wq + o_row * CDIM + c0 + (hi ? 8 : 0);
        A.q[0] = *(const uint4*)(ap);
        A.q[1] = *(const uint4*)(ap + 16);
#pragma unroll
        for (int t = 0; t < 4; ++t) {
            const int n = n0 + t * 16 + l15;
            const unsigned short* bp =
                y_t + ((size_t)(b * HW + n)) * CDIM + c0 + (hi ? 8 : 0);
            BF16x16 Bm;
            Bm.q[0] = *(const uint4*)(bp);
            Bm.q[1] = *(const uint4*)(bp + 16);
            acc[t] = wmma_bf16(A.v, Bm.v, acc[t]);
        }
    }

    const int o0 = mtile * 16;
    if (o0 < 384) {  // q or k: store [b][o][n]
#pragma unroll
        for (int t = 0; t < 4; ++t) {
            const int n = n0 + t * 16 + l15;
#pragma unroll
            for (int r = 0; r < 8; ++r) {
                const int o = o0 + hi * 8 + r;
                qk[((size_t)(b * 384 + o) << 14) + n] = f32_to_bf16(acc[t][r]);
            }
        }
    } else {         // v: store transposed v_t[b][n][c], 16B packed per lane
        const int ov = o0 - 384;
#pragma unroll
        for (int t = 0; t < 4; ++t) {
            const int n = n0 + t * 16 + l15;
            *(uint4*)(v_t + ((size_t)(b * HW + n)) * CDIM + ov + hi * 8) =
                pack8_bf16(acc[t]);
        }
    }
}

// --------------------------------------------- row sum-of-squares for F.normalize
__global__ void k_rowsumsq(const unsigned short* __restrict__ qk,
                           float* __restrict__ sumsq) {
    const int row = blockIdx.x;                 // 0 .. 8*384-1
    const unsigned short* p = qk + ((size_t)row << 14);
    float s = 0.f;
    for (int i = threadIdx.x; i < HW; i += 256) {
        float v = bf16_to_f32(p[i]);
        s += v * v;
    }
    __shared__ float red[256];
    red[threadIdx.x] = s;
    __syncthreads();
    for (int st = 128; st > 0; st >>= 1) {
        if (threadIdx.x < st) red[threadIdx.x] += red[threadIdx.x + st];
        __syncthreads();
    }
    if (threadIdx.x == 0) sumsq[row] = red[0];
}

__global__ void k_zero(float* __restrict__ p, int n) {
    int i = blockIdx.x * 256 + threadIdx.x;
    if (i < n) p[i] = 0.f;
}

// -------------------------------- Gram GEMM: attn_raw[b,h] = q(24xHW) @ k(24xHW)^T
// K=16384 split over 16 waves per (b,h); padded to 32x32 tiles; fp32 atomics.
__global__ void k_gemm_attn(const unsigned short* __restrict__ qk,
                            float* __restrict__ attn_raw) {
    const int wid  = blockIdx.x * 8 + (threadIdx.x >> 5);
    const int lane = threadIdx.x & 31;
    const int ks = wid & 15;
    const int bh = wid >> 4;                    // 0..63
    const int b = bh >> 3, h = bh & 7;
    const int hi = lane >> 4, l15 = lane & 15;

    v8f acc[2][2];
#pragma unroll
    for (int mi = 0; mi < 2; ++mi)
#pragma unroll
        for (int ni = 0; ni < 2; ++ni) acc[mi][ni] = v8f_zero();

    for (int step = 0; step < 32; ++step) {
        const int n0 = ks * 1024 + step * 32;
        BF16x16 A[2], Bm[2];
#pragma unroll
        for (int mi = 0; mi < 2; ++mi) {
            const int row = mi * 16 + l15;
            if (row < CT) {
                const unsigned short* ap =
                    qk + ((size_t)(b * 384 + h * CT + row) << 14) + n0 + (hi ? 8 : 0);
                A[mi].q[0] = *(const uint4*)(ap);
                A[mi].q[1] = *(const uint4*)(ap + 16);
            } else {
                A[mi].q[0] = uint4{0u,0u,0u,0u};
                A[mi].q[1] = uint4{0u,0u,0u,0u};
            }
        }
#pragma unroll
        for (int ni = 0; ni < 2; ++ni) {
            const int d = ni * 16 + l15;
            if (d < CT) {
                const unsigned short* bp =
                    qk + ((size_t)(b * 384 + 192 + h * CT + d) << 14) + n0 + (hi ? 8 : 0);
                Bm[ni].q[0] = *(const uint4*)(bp);
                Bm[ni].q[1] = *(const uint4*)(bp + 16);
            } else {
                Bm[ni].q[0] = uint4{0u,0u,0u,0u};
                Bm[ni].q[1] = uint4{0u,0u,0u,0u};
            }
        }
#pragma unroll
        for (int mi = 0; mi < 2; ++mi)
#pragma unroll
            for (int ni = 0; ni < 2; ++ni)
                acc[mi][ni] = wmma_bf16(A[mi].v, Bm[ni].v, acc[mi][ni]);
    }

#pragma unroll
    for (int mi = 0; mi < 2; ++mi)
#pragma unroll
        for (int ni = 0; ni < 2; ++ni) {
            const int d = ni * 16 + l15;
            if (d >= CT) continue;
#pragma unroll
            for (int r = 0; r < 8; ++r) {
                const int c = mi * 16 + hi * 8 + r;
                if (c < CT)
                    atomicAdd(&attn_raw[((size_t)bh * CT + c) * CT + d], acc[mi][ni][r]);
            }
        }
}

// ----------------------- scale by temperature / (||q|| * ||k||), softmax, pad 32x32
__global__ void k_softmax(const float* __restrict__ attn_raw,
                          const float* __restrict__ sumsq,
                          const float* __restrict__ temperature,
                          unsigned short* __restrict__ attn_pad) {
    const int bh = blockIdx.x;
    const int b = bh >> 3, h = bh & 7;
    const int j = threadIdx.x;                  // 0..31 (row)
    unsigned short* outp = attn_pad + (size_t)bh * 1024 + j * 32;
    if (j >= CT) {
        for (int d = 0; d < 32; ++d) outp[d] = 0;
        return;
    }
    const float t = temperature[h];
    const float inq = 1.f / fmaxf(sqrtf(sumsq[b * 384 + h * CT + j]), EPSN);
    float vals[CT];
    float m = -1e30f;
#pragma unroll
    for (int d = 0; d < CT; ++d) {
        const float ink = 1.f / fmaxf(sqrtf(sumsq[b * 384 + 192 + h * CT + d]), EPSN);
        float v = attn_raw[((size_t)bh * CT + j) * CT + d] * inq * ink * t;
        vals[d] = v;
        m = fmaxf(m, v);
    }
    float s = 0.f;
#pragma unroll
    for (int d = 0; d < CT; ++d) { vals[d] = __expf(vals[d] - m); s += vals[d]; }
    const float inv = 1.f / s;
#pragma unroll
    for (int d = 0; d < CT; ++d) outp[d] = f32_to_bf16(vals[d] * inv);
    for (int d = CT; d < 32; ++d) outp[d] = 0;
}

// --------------------------- out = attn(24x24, padded 32x32) @ v -> out_t[b][n][c]
__global__ void k_gemm_out(const unsigned short* __restrict__ attn_pad,
                           const unsigned short* __restrict__ v_t,
                           unsigned short* __restrict__ out_t) {
    const int wid  = blockIdx.x * 8 + (threadIdx.x >> 5);
    const int lane = threadIdx.x & 31;
    const int hi = lane >> 4, l15 = lane & 15;
    const int n64 = wid & 255;
    const int bh = wid >> 8;
    const int b = bh >> 3, h = bh & 7;
    const int n0 = n64 * 64;

    BF16x16 A[2];
#pragma unroll
    for (int mi = 0; mi < 2; ++mi) {
        const unsigned short* ap =
            attn_pad + (size_t)bh * 1024 + (mi * 16 + l15) * 32 + (hi ? 8 : 0);
        A[mi].q[0] = *(const uint4*)(ap);
        A[mi].q[1] = *(const uint4*)(ap + 16);
    }
#pragma unroll
    for (int t = 0; t < 4; ++t) {
        const int n = n0 + t * 16 + l15;
        const unsigned short* bp =
            v_t + ((size_t)(b * HW + n)) * CDIM + h * CT + (hi ? 8 : 0);
        BF16x16 Bm;
        Bm.q[0] = *(const uint4*)(bp);
        Bm.q[1] = *(const uint4*)(bp + 16);
        v8f a0 = v8f_zero(), a1 = v8f_zero();
        a0 = wmma_bf16(A[0].v, Bm.v, a0);       // rows c = 0..15
        a1 = wmma_bf16(A[1].v, Bm.v, a1);       // rows c = 16..31 (>=24 garbage)
        unsigned short* op = out_t + ((size_t)(b * HW + n)) * CDIM + h * CT;
        *(uint4*)(op + hi * 8) = pack8_bf16(a0);
        if (!hi) *(uint4*)(op + 16) = pack8_bf16(a1);   // rows 16..23 only
    }
}

// ------------------------------------------- final 1x1 proj: [192x192]x[192xN], f32 out
__global__ void k_gemm_proj(const unsigned short* __restrict__ wp,
                            const unsigned short* __restrict__ out_t,
                            float* __restrict__ out) {
    const int wid  = blockIdx.x * 8 + (threadIdx.x >> 5);
    const int lane = threadIdx.x & 31;
    const int hi = lane >> 4, l15 = lane & 15;
    const int n64   = wid & 255;
    const int mtile = (wid >> 8) % 12;
    const int b     = wid / (256 * 12);
    const int o_row = mtile * 16 + l15;
    const int n0 = n64 * 64;

    v8f acc[4];
#pragma unroll
    for (int t = 0; t < 4; ++t) acc[t] = v8f_zero();

    for (int kk = 0; kk < 6; ++kk) {
        const int c0 = kk * 32;
        BF16x16 A;
        const unsigned short* ap = wp + o_row * CDIM + c0 + (hi ? 8 : 0);
        A.q[0] = *(const uint4*)(ap);
        A.q[1] = *(const uint4*)(ap + 16);
#pragma unroll
        for (int t = 0; t < 4; ++t) {
            const int n = n0 + t * 16 + l15;
            const unsigned short* bp =
                out_t + ((size_t)(b * HW + n)) * CDIM + c0 + (hi ? 8 : 0);
            BF16x16 Bm;
            Bm.q[0] = *(const uint4*)(bp);
            Bm.q[1] = *(const uint4*)(bp + 16);
            acc[t] = wmma_bf16(A.v, Bm.v, acc[t]);
        }
    }
#pragma unroll
    for (int t = 0; t < 4; ++t) {
        const int n = n0 + t * 16 + l15;
#pragma unroll
        for (int r = 0; r < 8; ++r) {
            const int o = mtile * 16 + hi * 8 + r;
            out[((size_t)(b * CDIM + o) << 14) + n] = acc[t][r];
        }
    }
}

// ================================================================ host launcher
extern "C" void kernel_launch(void* const* d_in, const int* in_sizes, int n_in,
                              void* d_out, int out_size, void* d_ws, size_t ws_size,
                              hipStream_t stream) {
    (void)in_sizes; (void)n_in; (void)out_size; (void)ws_size;
    const float* x      = (const float*)d_in[0];
    const float* dw_w   = (const float*)d_in[1];
    const float* qkv_w  = (const float*)d_in[2];
    const float* proj_w = (const float*)d_in[3];
    const float* temp   = (const float*)d_in[4];

    // Workspace carve-up (~202 MB total; y_t reused as out_t)
    char* ws = (char*)d_ws;
    size_t off = 0;
    auto alloc = [&](size_t bytes) -> char* {
        char* p = ws + off;
        off = (off + bytes + 255) & ~(size_t)255;
        return p;
    };
    unsigned short* y_t      = (unsigned short*)alloc((size_t)BATCH * HW * CDIM * 2);       // 48 MiB (reused as out_t)
    unsigned short* qk       = (unsigned short*)alloc((size_t)BATCH * 384 * HW * 2);        // 96 MiB
    unsigned short* v_t      = (unsigned short*)alloc((size_t)BATCH * HW * CDIM * 2 + 256); // 48 MiB + guard
    unsigned short* wq       = (unsigned short*)alloc((size_t)OC3 * CDIM * 2);
    unsigned short* wp       = (unsigned short*)alloc((size_t)CDIM * CDIM * 2);
    float*          sumsq    = (float*)alloc((size_t)BATCH * 384 * 4);
    float*          attn_raw = (float*)alloc((size_t)64 * CT * CT * 4);
    unsigned short* attn_pad = (unsigned short*)alloc((size_t)64 * 32 * 32 * 2);

    k_convert_w<<<(OC3 * CDIM + 255) / 256, 256, 0, stream>>>(qkv_w, proj_w, wq, wp);
    k_dwconv<<<dim3(256, 6, BATCH), 256, 0, stream>>>(x, dw_w, y_t);
    k_gemm_qkv<<<(BATCH * 36 * 256) / 8, 256, 0, stream>>>(wq, y_t, qk, v_t);
    k_rowsumsq<<<BATCH * 384, 256, 0, stream>>>(qk, sumsq);
    k_zero<<<(64 * CT * CT + 255) / 256, 256, 0, stream>>>(attn_raw, 64 * CT * CT);
    k_gemm_attn<<<(64 * 16) / 8, 256, 0, stream>>>(qk, attn_raw);
    k_softmax<<<64, 32, 0, stream>>>(attn_raw, sumsq, temp, attn_pad);
    k_gemm_out<<<(64 * 256) / 8, 256, 0, stream>>>(attn_pad, v_t, y_t /* out_t */);
    k_gemm_proj<<<(BATCH * 12 * 256) / 8, 256, 0, stream>>>(wp, y_t, (float*)d_out);
}
